// CharLSTM_5342939317008
// MI455X (gfx1250) — matrix operand order repaired and checked by reading, not compiled, not run on hardware
//
#include <hip/hip_runtime.h>
#include <hip/hip_fp16.h>

typedef _Float16 v16h __attribute__((ext_vector_type(16)));
typedef _Float16 v8h  __attribute__((ext_vector_type(8)));
typedef float    v8f  __attribute__((ext_vector_type(8)));

#define VOCAB 256
#define EMB   32
#define HID   128
#define BATCH 512
#define TSTEPS 512
#define KTOT  160     // EMB + HID (combined contraction)
#define NKB   5       // KTOT / 32
#define WCP   168     // padded row length (halves) for combined gate weights [512][KTOT]
#define WCLSP 136     // padded row length (halves) for classifier weights [256][HID]
#define AP    168     // padded row length (halves) for A-tile [16][KTOT]

// gfx1250 has native V_TANH_F32 (transcendental).
__device__ __forceinline__ float fast_tanh(float x) {
#if __has_builtin(__builtin_amdgcn_tanhf)
    return __builtin_amdgcn_tanhf(x);
#else
    return tanhf(x);
#endif
}
__device__ __forceinline__ float fast_sigmoid(float x) {
    return __builtin_fmaf(fast_tanh(0.5f * x), 0.5f, 0.5f);
}

// Build a v16h A/B fragment from two 16-byte LDS chunks (each -> ds_load_b128).
__device__ __forceinline__ v16h frag_from(const _Float16* p0, const _Float16* p1) {
    v8h lo = *(const v8h*)p0;
    v8h hi = *(const v8h*)p1;
    return __builtin_shufflevector(lo, hi, 0,1,2,3,4,5,6,7,8,9,10,11,12,13,14,15);
}

__device__ __forceinline__ v8f splat8(float x) {
    v8f v = {x, x, x, x, x, x, x, x};
    return v;
}

__global__ __launch_bounds__(256)
void charlstm_fused(const long long* __restrict__ tok,   // [B, T] int64
                    const float* __restrict__ embf,      // [VOCAB, EMB]
                    const float* __restrict__ Wih,       // [4H, EMB]
                    const float* __restrict__ Whh,       // [4H, HID]
                    const float* __restrict__ bih,       // [4H]
                    const float* __restrict__ bhh,       // [4H]
                    const float* __restrict__ Wcls,      // [VOCAB, HID]
                    const float* __restrict__ bcls,      // [VOCAB]
                    float* __restrict__ out,             // [B, VOCAB]
                    int T)                               // must be even
{
    extern __shared__ char smem[];
    _Float16* sWc   = (_Float16*)smem;            // [512][WCP]   gate weights f16 (W_ih | W_hh)
    _Float16* sWcls = sWc   + 512 * WCP;          // [256][WCLSP] classifier weights f16
    _Float16* sEmb  = sWcls + 256 * WCLSP;        // [256][EMB]   embedding table f16
    _Float16* sA0   = sEmb  + VOCAB * EMB;        // [16][AP]     ping A-tile (x_t | h_{t-1})
    _Float16* sA1   = sA0   + 16 * AP;            // [16][AP]     pong A-tile
    float*    sBias = (float*)(sA1 + 16 * AP);    // [512]  b_ih + b_hh
    float*    sBcls = sBias + 512;                // [256]

    const int tid   = threadIdx.x;
    const int lane  = tid & 31;
    const int wave  = tid >> 5;    // 0..7, owns hidden cols [wave*16, wave*16+16)
    const int mrow  = lane & 15;   // M for A-frag / N (column) for B-frag
    const int hs    = lane >> 4;   // lane-half select per ISA fragment layout
    const int bbase = blockIdx.x * 16;

    // ---------------- preload & f32->f16 convert into LDS ----------------
    for (int i = tid; i < 512 * EMB; i += 256)
        sWc[(i / EMB) * WCP + (i % EMB)] = (_Float16)Wih[i];
    for (int i = tid; i < 512 * HID; i += 256)
        sWc[(i / HID) * WCP + EMB + (i % HID)] = (_Float16)Whh[i];
    for (int i = tid; i < VOCAB * HID; i += 256)
        sWcls[(i / HID) * WCLSP + (i % HID)] = (_Float16)Wcls[i];
    for (int i = tid; i < VOCAB * EMB; i += 256)
        sEmb[i] = (_Float16)embf[i];
    for (int i = tid; i < 512; i += 256) sBias[i] = bih[i] + bhh[i];
    for (int i = tid; i < VOCAB; i += 256) sBcls[i] = bcls[i];
    for (int i = tid; i < 16 * AP; i += 256) sA0[i] = (_Float16)0.0f;  // h0 = 0 (+x area)
    __syncthreads();

    // x_0 gather into ping buffer
    if (tid < 16) {
        long long tk = tok[(long long)(bbase + tid) * T];
        const _Float16* src = sEmb + tk * EMB;
        _Float16* dst = sA0 + tid * AP;
        #pragma unroll
        for (int e = 0; e < EMB; e += 8)
            *(v8h*)(dst + e) = *(const v8h*)(src + e);
    }

    const int nb = wave * 16;  // this wave's hidden-column base
    // Per-lane gate biases are time-invariant: hoist; they seed the WMMA C operand.
    const float gb0 = sBias[0 * HID + nb + mrow];   // i
    const float gb1 = sBias[1 * HID + nb + mrow];   // f
    const float gb2 = sBias[2 * HID + nb + mrow];   // g
    const float gb3 = sBias[3 * HID + nb + mrow];   // o

    v8f c = {};  // cell state for this wave's 16 hidden cols (C-tile layout)

    __syncthreads();  // x_0 visible

    // One LSTM timestep: read (x_t | h_{t-1}) from sAr, write (x_{t+1} | h_t) to sAw.
    // Fixed buffer pointers per call site -> loop-invariant LDS addresses.
    auto lstm_step = [&](const _Float16* sAr, _Float16* sAw, int t) {
        // A fragments for all 5 K-blocks (shared by this wave's 4 gate tiles)
        v16h afrag[NKB];
        #pragma unroll
        for (int kb = 0; kb < NKB; ++kb) {
            const _Float16* ap = sAr + mrow * AP + kb * 32 + hs * 8;
            afrag[kb] = frag_from(ap, ap + 16);
        }

        // 4 gate tiles: gate G in {i,f,g,o} -> N-tile G*8 + wave.
        // Accumulators seeded with the gate bias (D = A*B + C).
        v8f acc[4];
        acc[0] = splat8(gb0); acc[1] = splat8(gb1);
        acc[2] = splat8(gb2); acc[3] = splat8(gb3);
        #pragma unroll
        for (int G = 0; G < 4; ++G) {
            const int ncol = (G * 8 + wave) * 16 + mrow;  // row of [W_ih|W_hh]
            v8f a_ = acc[G];
            #pragma unroll
            for (int kb = 0; kb < NKB; ++kb) {
                const _Float16* bp = sWc + ncol * WCP + kb * 32 + hs * 8;
                v16h bfrag = frag_from(bp, bp + 16);
                a_ = __builtin_amdgcn_wmma_f32_16x16x32_f16(
                        false, afrag[kb], false, bfrag, (short)0, a_, false, false);
            }
            acc[G] = a_;
        }

        // Elementwise LSTM cell update; h_t (f16) into the *other* A buffer
        #pragma unroll
        for (int r = 0; r < 8; ++r) {
            const float iv = fast_sigmoid(acc[0][r]);
            const float fv = fast_sigmoid(acc[1][r]);
            const float gv = fast_tanh   (acc[2][r]);
            const float ov = fast_sigmoid(acc[3][r]);
            const float cn = fv * c[r] + iv * gv;
            c[r] = cn;
            const float hv = ov * fast_tanh(cn);
            const int M = r + hs * 8;                // C-layout row
            sAw[M * AP + EMB + nb + mrow] = (_Float16)hv;
        }

        // Gather x_{t+1} into the write buffer; prefetch token stream ahead
        if (t + 1 < T && tid < 16) {
            const long long* tp = tok + (long long)(bbase + tid) * T + (t + 1);
            __builtin_prefetch(tp + 8, 0, 0);
            long long tk = *tp;
            const _Float16* src = sEmb + tk * EMB;
            _Float16* dst = sAw + tid * AP;
            #pragma unroll
            for (int e = 0; e < EMB; e += 8)
                *(v8h*)(dst + e) = *(const v8h*)(src + e);
        }

        __syncthreads();  // single barrier per timestep
    };

    // Manual 2x unroll: both bodies get compile-time-fixed ping/pong pointers.
    for (int t = 0; t < T; t += 2) {
        lstm_step(sA0, sA1, t);
        lstm_step(sA1, sA0, t + 1);
    }

    // T even -> h_T is in sA0
    const _Float16* sAfin = sA0;

    // ---------------- classifier: out = h_T @ W_cls^T + b_cls ----------------
    #pragma unroll
    for (int q = 0; q < 2; ++q) {
        const int tn   = wave * 2 + q;       // 16 vocab tiles / 8 waves
        const int ncol = tn * 16 + mrow;     // vocab index
        v8f a_ = splat8(sBcls[ncol]);        // seed with classifier bias
        #pragma unroll
        for (int kb = 0; kb < 4; ++kb) {     // K = HID = 128 (A-tile cols 32..159)
            const _Float16* ap = sAfin + mrow * AP + EMB + kb * 32 + hs * 8;
            v16h afrag = frag_from(ap, ap + 16);
            const _Float16* bp = sWcls + ncol * WCLSP + kb * 32 + hs * 8;
            v16h bfrag = frag_from(bp, bp + 16);
            a_ = __builtin_amdgcn_wmma_f32_16x16x32_f16(
                    false, afrag, false, bfrag, (short)0, a_, false, false);
        }
        #pragma unroll
        for (int r = 0; r < 8; ++r) {
            const int M = r + hs * 8;
            out[(long long)(bbase + M) * VOCAB + ncol] = a_[r];
        }
    }
}

extern "C" void kernel_launch(void* const* d_in, const int* in_sizes, int n_in,
                              void* d_out, int out_size, void* d_ws, size_t ws_size,
                              hipStream_t stream) {
    (void)in_sizes; (void)n_in; (void)d_ws; (void)ws_size; (void)out_size;

    const long long* tok  = (const long long*)d_in[0];
    const float*     embf = (const float*)d_in[1];
    const float*     Wih  = (const float*)d_in[2];
    const float*     Whh  = (const float*)d_in[3];
    const float*     bih  = (const float*)d_in[4];
    const float*     bhh  = (const float*)d_in[5];
    const float*     Wcls = (const float*)d_in[6];
    const float*     bcls = (const float*)d_in[7];
    float*           out  = (float*)d_out;

    const size_t smem_bytes =
        sizeof(_Float16) * (512 * WCP + 256 * WCLSP + VOCAB * EMB + 2 * 16 * AP) +
        sizeof(float) * (512 + VOCAB);   // ~272 KB, within 320 KB/WGP

    // Opt in to large dynamic LDS (idempotent, deterministic).
    hipFuncSetAttribute((const void*)charlstm_fused,
                        hipFuncAttributeMaxDynamicSharedMemorySize,
                        (int)smem_bytes);

    dim3 grid(BATCH / 16);   // 32 workgroups, each owns a 16-row batch tile
    dim3 block(256);         // 8 waves: 2 per SIMD32 for latency hiding
    charlstm_fused<<<grid, block, smem_bytes, stream>>>(
        tok, embf, Wih, Whh, bih, bhh, Wcls, bcls, out, TSTEPS);
}
